// CrossGatedAttentionGate_51668456570919
// MI455X (gfx1250) — compile-verified, matrix operand order checked by reading, not run on hardware
//
#include <hip/hip_runtime.h>
#include <hip/hip_bf16.h>

// ---------------------------------------------------------------------------
// CDNA5 / gfx1250 implementation of CrossGatedAttentionGate.
// Dense layers: V_WMMA_F32_16X16X32_BF16, wave32, block tile 128x64,
// each wave owns a 32x32 output slab (4 WMMAs per K-step between barriers).
// Mamba scan: chunk-parallel 3-phase scan (32768 workers instead of 2048).
// ---------------------------------------------------------------------------

typedef __attribute__((ext_vector_type(16))) __bf16        v16bf;
typedef __attribute__((ext_vector_type(8)))  float         v8f;
typedef __attribute__((ext_vector_type(4)))  unsigned int  v4u;

#define DEV __device__ __forceinline__

DEV unsigned short f2bf(float f) {          // RNE f32 -> bf16
    unsigned int u = __float_as_uint(f);
    u += 0x7FFFu + ((u >> 16) & 1u);
    return (unsigned short)(u >> 16);
}
DEV float bf2f(unsigned short u) { return __uint_as_float(((unsigned int)u) << 16); }
DEV float sigm(float v)  { return 1.0f / (1.0f + __expf(-v)); }
DEV float siluf(float v) { return v * sigm(v); }

// Problem constants
static constexpr int Bq   = 2;
static constexpr int Lh   = 4096;      // H*W
static constexpr int Mrow = Bq * Lh;   // 8192 rows for every GEMM
static constexpr int NI   = 8;         // mamba instances
static constexpr int CDIM = 64;
static constexpr int DIN  = 128;
static constexpr int NCH  = 16;        // scan chunks per sequence
static constexpr int CLEN = Lh / NCH;  // 256 steps per chunk
static constexpr int NCHAIN = NI * Bq * DIN;   // 2048 scan chains

#define LDS_STR 40   // LDS row stride in ushort (80B: 16B-aligned, de-conflicted)

// ---------------------------------------------------------------------------
// WMMA bf16 GEMM:  Out[m][n] = epilogue( sum_k A[m][k] * Bw[n][k] )
// A: f32 [inst][M][K] row-major.  Bw: f32 [inst][N][K] row-major (weights).
// 256 threads = 8 waves; block tile 128(M) x 64(N); K step 32.
// wave (wm=wid&3, wn=wid>>2) owns rows wm*32..+31, cols wn*32..+31 -> 4 WMMA.
// ---------------------------------------------------------------------------
__global__ __launch_bounds__(256) void gemm_wmma_bf16(
    const float* __restrict__ A,  long sAi,
    const float* __restrict__ Bw, long sBi,
    void* __restrict__ Out,       long sOi,
    int N, int K,
    const float* __restrict__ bias, const float* __restrict__ scale,
    const float* __restrict__ shift, long sEi,
    int do_relu, int out_bf16)
{
    __shared__ __align__(16) unsigned short sA[128][LDS_STR];
    __shared__ __align__(16) unsigned short sB[64][LDS_STR];

    const int inst = blockIdx.z;
    const float* Ai = A  + (long)inst * sAi + (long)blockIdx.y * 128 * K;
    const float* Bi = Bw + (long)inst * sBi;
    const int n0   = blockIdx.x * 64;
    const int tid  = threadIdx.x;
    const int lane = tid & 31, wid = tid >> 5;
    const int wm   = wid & 3,  wn  = wid >> 2;
    const int half = lane >> 4, lr = lane & 15;

    v8f acc00 = {0.f,0.f,0.f,0.f,0.f,0.f,0.f,0.f};
    v8f acc01 = acc00, acc10 = acc00, acc11 = acc00;

    const int rA = tid >> 1;          // 0..127 A staging row
    const int kA = (tid & 1) * 16;    // 0 or 16
    const int rB = tid >> 2;          // 0..63  B staging row
    const int kB = (tid & 3) * 8;     // 0,8,16,24

    for (int k0 = 0; k0 < K; k0 += 32) {
        { // stage A 128x32
            const float* src = Ai + (long)rA * K + k0 + kA;
            #pragma unroll
            for (int j = 0; j < 16; ++j) sA[rA][kA + j] = f2bf(src[j]);
        }
        { // stage B 64x32 (zero-fill n >= N, e.g. N=36 x-proj)
            int n = n0 + rB;
            if (n < N) {
                const float* src = Bi + (long)n * K + k0 + kB;
                #pragma unroll
                for (int j = 0; j < 8; ++j) sB[rB][kB + j] = f2bf(src[j]);
            } else {
                #pragma unroll
                for (int j = 0; j < 8; ++j) sB[rB][kB + j] = 0;
            }
        }
        __syncthreads();

        // fragments per ISA 7.12.2 (16-bit A 16x32): lane half selects K +8,
        // elements j<8 -> K = j + 8*half, j>=8 -> K = 16 + (j-8) + 8*half.
        union { v16bf v; v4u u[2]; } fa0, fa1, fb0, fb1;
        {
            int ra = wm * 32 + lr;
            fa0.u[0] = *(const v4u*)&sA[ra][half * 8];
            fa0.u[1] = *(const v4u*)&sA[ra][16 + half * 8];
            fa1.u[0] = *(const v4u*)&sA[ra + 16][half * 8];
            fa1.u[1] = *(const v4u*)&sA[ra + 16][16 + half * 8];
        }
        {
            int rb = wn * 32 + lr;
            fb0.u[0] = *(const v4u*)&sB[rb][half * 8];
            fb0.u[1] = *(const v4u*)&sB[rb][16 + half * 8];
            fb1.u[0] = *(const v4u*)&sB[rb + 16][half * 8];
            fb1.u[1] = *(const v4u*)&sB[rb + 16][16 + half * 8];
        }
        acc00 = __builtin_amdgcn_wmma_f32_16x16x32_bf16(false, fa0.v, false, fb0.v,
                                                        (short)0, acc00, false, false);
        acc01 = __builtin_amdgcn_wmma_f32_16x16x32_bf16(false, fa0.v, false, fb1.v,
                                                        (short)0, acc01, false, false);
        acc10 = __builtin_amdgcn_wmma_f32_16x16x32_bf16(false, fa1.v, false, fb0.v,
                                                        (short)0, acc10, false, false);
        acc11 = __builtin_amdgcn_wmma_f32_16x16x32_bf16(false, fa1.v, false, fb1.v,
                                                        (short)0, acc11, false, false);
        __syncthreads();
    }

    // epilogue + store.  C/D layout: VGPR v -> M = v + 8*half, col = lane%16.
    const float* bi = bias  ? bias  + (long)inst * sEi : nullptr;
    const float* si = scale ? scale + (long)inst * sEi : nullptr;
    const float* ti = shift ? shift + (long)inst * sEi : nullptr;
    const long mBase = (long)blockIdx.y * 128 + wm * 32 + 8 * half;
    #pragma unroll
    for (int sm = 0; sm < 2; ++sm) {
        #pragma unroll
        for (int v = 0; v < 8; ++v) {
            long m = mBase + sm * 16 + v;
            #pragma unroll
            for (int sn = 0; sn < 2; ++sn) {
                int n = n0 + wn * 32 + sn * 16 + lr;
                if (n >= N) continue;
                float val = sm ? (sn ? acc11[v] : acc10[v])
                               : (sn ? acc01[v] : acc00[v]);
                if (bi) val += bi[n];
                if (si) val = val * si[n] + ti[n];
                if (do_relu) val = fmaxf(val, 0.f);
                if (out_bf16)
                    ((unsigned short*)Out)[(long)inst * sOi + m * N + n] = f2bf(val);
                else
                    ((float*)Out)[(long)inst * sOi + m * N + n] = val;
            }
        }
    }
}

// ---------------------------------------------------------------------------
// 3x3 conv 256->64 as implicit WMMA GEMM: K = 9 taps x 256 ic.
// Block tile 128(pixels) x 64(out-ch); same 32x32-per-wave scheme.
// ---------------------------------------------------------------------------
__global__ __launch_bounds__(256) void conv3_wmma_bf16(
    const float* __restrict__ cat2,   // [4][8192][64]
    const float* __restrict__ Wc,     // [64][256][3][3]
    float* __restrict__ Outp,         // [8192][64]
    const float* __restrict__ bias, const float* __restrict__ scale,
    const float* __restrict__ shift)
{
    __shared__ __align__(16) unsigned short sA[128][LDS_STR];
    __shared__ __align__(16) unsigned short sB[64][LDS_STR];

    const int m0   = blockIdx.y * 128;
    const int b    = m0 >> 12;
    const int tid  = threadIdx.x;
    const int lane = tid & 31, wid = tid >> 5;
    const int wm   = wid & 3,  wn  = wid >> 2;
    const int half = lane >> 4, lr = lane & 15;
    const int rA   = tid >> 1;            // 0..127
    const int kA   = (tid & 1) * 16;
    const int rB   = tid >> 2;            // 0..63
    const int kB   = (tid & 3) * 8;

    const int l  = (m0 + rA) & 4095;
    const int py = l >> 6, px = l & 63;

    v8f acc00 = {0.f,0.f,0.f,0.f,0.f,0.f,0.f,0.f};
    v8f acc01 = acc00, acc10 = acc00, acc11 = acc00;

    for (int tap = 0; tap < 9; ++tap) {
        const int dy = tap / 3 - 1, dx = tap % 3 - 1;
        const int yy = py + dy, xx = px + dx;
        const bool inb = ((unsigned)yy < 64u) && ((unsigned)xx < 64u);
        const long pix = (long)(b << 12) + ((yy & 63) << 6) + (xx & 63);

        for (int kc = 0; kc < 256; kc += 32) {
            #pragma unroll
            for (int j = 0; j < 16; ++j) {              // stage A
                int ic = kc + kA + j;
                float v = inb ? cat2[(((long)(ic >> 6) * Mrow) + pix) * 64 + (ic & 63)]
                              : 0.f;
                sA[rA][kA + j] = f2bf(v);
            }
            #pragma unroll
            for (int j = 0; j < 8; ++j) {               // stage B (n = rB)
                int ic = kc + kB + j;
                sB[rB][kB + j] = f2bf(Wc[(((long)rB * 256 + ic) * 3 + (dy + 1)) * 3 + (dx + 1)]);
            }
            __syncthreads();

            union { v16bf v; v4u u[2]; } fa0, fa1, fb0, fb1;
            int ra = wm * 32 + lr;
            fa0.u[0] = *(const v4u*)&sA[ra][half * 8];
            fa0.u[1] = *(const v4u*)&sA[ra][16 + half * 8];
            fa1.u[0] = *(const v4u*)&sA[ra + 16][half * 8];
            fa1.u[1] = *(const v4u*)&sA[ra + 16][16 + half * 8];
            int rb = wn * 32 + lr;
            fb0.u[0] = *(const v4u*)&sB[rb][half * 8];
            fb0.u[1] = *(const v4u*)&sB[rb][16 + half * 8];
            fb1.u[0] = *(const v4u*)&sB[rb + 16][half * 8];
            fb1.u[1] = *(const v4u*)&sB[rb + 16][16 + half * 8];

            acc00 = __builtin_amdgcn_wmma_f32_16x16x32_bf16(false, fa0.v, false, fb0.v,
                                                            (short)0, acc00, false, false);
            acc01 = __builtin_amdgcn_wmma_f32_16x16x32_bf16(false, fa0.v, false, fb1.v,
                                                            (short)0, acc01, false, false);
            acc10 = __builtin_amdgcn_wmma_f32_16x16x32_bf16(false, fa1.v, false, fb0.v,
                                                            (short)0, acc10, false, false);
            acc11 = __builtin_amdgcn_wmma_f32_16x16x32_bf16(false, fa1.v, false, fb1.v,
                                                            (short)0, acc11, false, false);
            __syncthreads();
        }
    }

    const long mBase = (long)m0 + wm * 32 + 8 * half;
    #pragma unroll
    for (int sm = 0; sm < 2; ++sm) {
        #pragma unroll
        for (int v = 0; v < 8; ++v) {
            long m = mBase + sm * 16 + v;
            #pragma unroll
            for (int sn = 0; sn < 2; ++sn) {
                int n = wn * 32 + sn * 16 + lr;
                float val = (sm ? (sn ? acc11[v] : acc10[v])
                                : (sn ? acc01[v] : acc00[v])) + bias[n];
                Outp[m * 64 + n] = fmaxf(val * scale[n] + shift[n], 0.f);
            }
        }
    }
}

// ---------------------------------------------------------------------------
// Elementwise / scan kernels
// ---------------------------------------------------------------------------
__global__ void pack_seqs(const float* __restrict__ g, const float* __restrict__ x,
                          float* __restrict__ seqs)     // [8][8192][64]
{
    long t = (long)blockIdx.x * blockDim.x + threadIdx.x;
    if (t >= (long)NI * Mrow * CDIM) return;
    int c = t & 63;
    long rr = t >> 6;
    int l = rr & 4095;  rr >>= 12;
    int b = rr & 1;     int i = (int)(rr >> 1);
    const float* src = (i < 4) ? g : x;
    int f = (i & 3) * 64 + c;
    seqs[t] = src[((long)b * 256 + f) * Lh + l];
}

__global__ void dwconv1d_silu(const unsigned short* __restrict__ xz,   // [8][8192][256] bf16
                              const float* __restrict__ cw,            // [8][128][4]
                              const float* __restrict__ cb,            // [8][128]
                              float* __restrict__ xc)                  // [8][8192][128]
{
    long t = (long)blockIdx.x * blockDim.x + threadIdx.x;
    if (t >= (long)NI * Mrow * DIN) return;
    int d = t & 127;
    long rr = t >> 7;
    int l = rr & 4095;  rr >>= 12;
    int b = rr & 1;     int i = (int)(rr >> 1);
    const float* w = cw + ((long)i * DIN + d) * 4;
    float s = cb[i * DIN + d];
    long rowb = (long)i * Mrow + b * Lh;
    #pragma unroll
    for (int k = 0; k < 4; ++k) {
        int lk = l + k - 3;
        if (lk >= 0) s += bf2f(xz[(rowb + lk) * 256 + d]) * w[k];
    }
    xc[t] = siluf(s);
}

// Chunk-parallel selective scan.
// chain = (i*2+b)*128 + d (2048 chains); NCH=16 chunks of CLEN=256 steps.
// pass1: per (chain,chunk) compute transition Prod[n]=prod dA, Hend with h0=0.
// combine: per (chain,n) serial stitch over 16 chunks -> Hstart.
// pass2: per (chain,chunk) rerun with correct h0, emit gated y.
DEV void scan_decode(int t, int& i, int& b, int& d, int& chunk) {
    d = t & 127; chunk = (t >> 7) & 15; b = (t >> 11) & 1; i = t >> 12;
}

__global__ void scan_pass1(const float* __restrict__ dbl,             // [8][8192][36]
                           const float* __restrict__ xc,              // [8][8192][128]
                           const float* __restrict__ dtw, const float* __restrict__ dtb,
                           const float* __restrict__ Alog,
                           float* __restrict__ Prod,                  // [2048][16][16]
                           float* __restrict__ Hend)                  // [2048][16][16]
{
    int t = blockIdx.x * blockDim.x + threadIdx.x;
    if (t >= NCHAIN * NCH) return;
    int i, b, d, chunk; scan_decode(t, i, b, d, chunk);

    float Ad[16];
    #pragma unroll
    for (int n = 0; n < 16; ++n) Ad[n] = -__expf(Alog[((long)i * DIN + d) * 16 + n]);
    const float* wdt = dtw + ((long)i * DIN + d) * 4;
    const float w0 = wdt[0], w1 = wdt[1], w2 = wdt[2], w3 = wdt[3];
    const float bb = dtb[i * DIN + d];

    float h[16], P[16];
    #pragma unroll
    for (int n = 0; n < 16; ++n) { h[n] = 0.f; P[n] = 1.f; }

    const long rowb = (long)i * Mrow + b * Lh + chunk * CLEN;
    for (int l = 0; l < CLEN; ++l) {
        const long row = rowb + l;
        const float* db = dbl + row * 36;
        __builtin_prefetch(db + 36, 0, 1);               // -> global_prefetch_b8
        float dt = db[0] * w0 + db[1] * w1 + db[2] * w2 + db[3] * w3 + bb;
        dt = (dt > 20.f) ? dt : log1pf(__expf(dt));      // softplus
        float xv = xc[row * DIN + d];
        #pragma unroll
        for (int n = 0; n < 16; ++n) {
            float dA = __expf(dt * Ad[n]);
            P[n] *= dA;
            h[n] = h[n] * dA + dt * db[4 + n] * xv;
        }
    }
    const long chain = ((long)(i * 2 + b) * DIN + d);
    float* Pp = Prod + (chain * NCH + chunk) * 16;
    float* He = Hend + (chain * NCH + chunk) * 16;
    #pragma unroll
    for (int n = 0; n < 16; ++n) { Pp[n] = P[n]; He[n] = h[n]; }
}

__global__ void scan_combine(const float* __restrict__ Prod,
                             const float* __restrict__ Hend,
                             float* __restrict__ Hstart)              // [2048][16][16]
{
    int t = blockIdx.x * blockDim.x + threadIdx.x;
    if (t >= NCHAIN * 16) return;
    int n = t & 15;
    long chain = t >> 4;
    float h = 0.f;
    for (int c = 0; c < NCH; ++c) {
        long idx = (chain * NCH + c) * 16 + n;
        Hstart[idx] = h;
        h = Prod[idx] * h + Hend[idx];
    }
}

__global__ void scan_pass2(const float* __restrict__ dbl,
                           const float* __restrict__ xc,
                           const unsigned short* __restrict__ xz,     // z = cols 128..255
                           const float* __restrict__ dtw, const float* __restrict__ dtb,
                           const float* __restrict__ Alog, const float* __restrict__ Dpp,
                           const float* __restrict__ Hstart,
                           float* __restrict__ yb)                    // [8][8192][128]
{
    int t = blockIdx.x * blockDim.x + threadIdx.x;
    if (t >= NCHAIN * NCH) return;
    int i, b, d, chunk; scan_decode(t, i, b, d, chunk);

    float Ad[16];
    #pragma unroll
    for (int n = 0; n < 16; ++n) Ad[n] = -__expf(Alog[((long)i * DIN + d) * 16 + n]);
    const float* wdt = dtw + ((long)i * DIN + d) * 4;
    const float w0 = wdt[0], w1 = wdt[1], w2 = wdt[2], w3 = wdt[3];
    const float bb = dtb[i * DIN + d], dp = Dpp[i * DIN + d];

    const long chain = ((long)(i * 2 + b) * DIN + d);
    const float* hs = Hstart + (chain * NCH + chunk) * 16;
    float h[16];
    #pragma unroll
    for (int n = 0; n < 16; ++n) h[n] = hs[n];

    const long rowb = (long)i * Mrow + b * Lh + chunk * CLEN;
    for (int l = 0; l < CLEN; ++l) {
        const long row = rowb + l;
        const float* db = dbl + row * 36;
        __builtin_prefetch(db + 36, 0, 1);
        float dt = db[0] * w0 + db[1] * w1 + db[2] * w2 + db[3] * w3 + bb;
        dt = (dt > 20.f) ? dt : log1pf(__expf(dt));
        float xv = xc[row * DIN + d];
        float zv = bf2f(xz[row * 256 + 128 + d]);
        float acc = 0.f;
        #pragma unroll
        for (int n = 0; n < 16; ++n) {
            float dA = __expf(dt * Ad[n]);
            h[n] = h[n] * dA + dt * db[4 + n] * xv;
            acc += h[n] * db[20 + n];
        }
        yb[row * DIN + d] = (acc + dp * xv) * siluf(zv);
    }
}

DEV float dw3x3(const float* __restrict__ base, int y, int x, int c,
                const float* __restrict__ w9)
{
    float a = 0.f;
    #pragma unroll
    for (int ky = 0; ky < 3; ++ky) {
        int yy = y + ky - 1;
        if ((unsigned)yy >= 64u) continue;
        #pragma unroll
        for (int kx = 0; kx < 3; ++kx) {
            int xx = x + kx - 1;
            if ((unsigned)xx >= 64u) continue;
            a += base[(long)((yy << 6) + xx) * 64 + c] * w9[ky * 3 + kx];
        }
    }
    return a;
}

__global__ void branch_cross(const float* __restrict__ mbuf,          // [8][8192][64]
                             const float* __restrict__ gw, const float* __restrict__ gb,
                             const float* __restrict__ gs, const float* __restrict__ gt,
                             const float* __restrict__ xw, const float* __restrict__ xb,
                             const float* __restrict__ xs, const float* __restrict__ xt,
                             float* __restrict__ cross)               // [4][8192][64]
{
    long t = (long)blockIdx.x * blockDim.x + threadIdx.x;
    if (t >= 4L * Mrow * CDIM) return;
    int c = t & 63;
    long rr = t >> 6;
    int l = rr & 4095;  rr >>= 12;
    int b = rr & 1;     int br = (int)(rr >> 1);
    int y = l >> 6, x = l & 63;

    const float* gmb = mbuf + ((long)br * Mrow + b * Lh) * 64;
    const float* xmb = mbuf + ((long)(br + 4) * Mrow + b * Lh) * 64;
    int wc = br * 64 + c;
    float gc  = dw3x3(gmb, y, x, c, gw + (long)wc * 9) + gb[wc];
    float xcv = dw3x3(xmb, y, x, c, xw + (long)wc * 9) + xb[wc];
    gc  = fmaxf(gc  * gs[wc] + gt[wc], 0.f);
    xcv = fmaxf(xcv * xs[wc] + xt[wc], 0.f);
    float gmv = gmb[(long)l * 64 + c];
    float xmv = xmb[(long)l * 64 + c];
    cross[t] = xmv * sigm(gc) + gmv * sigm(xcv);
}

__global__ void branch_pdw(const float* __restrict__ cross,           // [4][8192][64]
                           const float* __restrict__ pw, const float* __restrict__ pb,
                           const float* __restrict__ ps, const float* __restrict__ pt,
                           float* __restrict__ h1)                    // [4][8192][64]
{
    long t = (long)blockIdx.x * blockDim.x + threadIdx.x;
    if (t >= 4L * Mrow * CDIM) return;
    int c = t & 63;
    long rr = t >> 6;
    int l = rr & 4095;  rr >>= 12;
    int b = rr & 1;     int br = (int)(rr >> 1);
    int y = l >> 6, x = l & 63;
    const float* base = cross + ((long)br * Mrow + b * Lh) * 64;
    int wc = br * 64 + c;
    float v = dw3x3(base, y, x, c, pw + (long)wc * 9) + pb[wc];
    h1[t] = fmaxf(v * ps[wc] + pt[wc], 0.f);
}

__global__ void psi_kernel(const float* __restrict__ h3,              // [8192][64]
                           const float* __restrict__ c1w, const float* __restrict__ c1b,
                           const float* __restrict__ c1s, const float* __restrict__ c1t,
                           float* __restrict__ psi)                   // [8192]
{
    int t = blockIdx.x * blockDim.x + threadIdx.x;
    if (t >= Mrow) return;
    float s = 0.f;
    const float* h = h3 + (long)t * 64;
    #pragma unroll
    for (int o = 0; o < 64; ++o) s += h[o] * c1w[o];
    s = (s + c1b[0]) * c1s[0] + c1t[0];
    psi[t] = sigm(s);
}

__global__ void final_mul(const float* __restrict__ x, const float* __restrict__ psi,
                          float* __restrict__ out)                    // [2][256][64][64]
{
    long t = (long)blockIdx.x * blockDim.x + threadIdx.x;
    if (t >= 2L * 256 * Lh) return;
    int l = t & 4095;
    int b = (int)(t >> 20);            // t / (256*4096)
    out[t] = x[t] * psi[b * Lh + l];
}

// ---------------------------------------------------------------------------
// Host launcher
// ---------------------------------------------------------------------------
extern "C" void kernel_launch(void* const* d_in, const int* in_sizes, int n_in,
                              void* d_out, int out_size, void* d_ws, size_t ws_size,
                              hipStream_t stream)
{
    const float* g    = (const float*)d_in[0];
    const float* x    = (const float*)d_in[1];
    const float* inw  = (const float*)d_in[2];
    const float* cw   = (const float*)d_in[3];
    const float* cb   = (const float*)d_in[4];
    const float* xpw  = (const float*)d_in[5];
    const float* dtw  = (const float*)d_in[6];
    const float* dtb  = (const float*)d_in[7];
    const float* Alog = (const float*)d_in[8];
    const float* Dpp  = (const float*)d_in[9];
    const float* ow   = (const float*)d_in[10];
    const float* dwg_w = (const float*)d_in[11];
    const float* dwg_b = (const float*)d_in[12];
    const float* dwg_s = (const float*)d_in[13];
    const float* dwg_t = (const float*)d_in[14];
    const float* dwx_w = (const float*)d_in[15];
    const float* dwx_b = (const float*)d_in[16];
    const float* dwx_s = (const float*)d_in[17];
    const float* dwx_t = (const float*)d_in[18];
    const float* pdw_w = (const float*)d_in[19];
    const float* pdw_b = (const float*)d_in[20];
    const float* pdw_s = (const float*)d_in[21];
    const float* pdw_t = (const float*)d_in[22];
    const float* p1_w  = (const float*)d_in[23];
    const float* p1_b  = (const float*)d_in[24];
    const float* p1_s  = (const float*)d_in[25];
    const float* p1_t  = (const float*)d_in[26];
    const float* c3_w  = (const float*)d_in[27];
    const float* c3_b  = (const float*)d_in[28];
    const float* c3_s  = (const float*)d_in[29];
    const float* c3_t  = (const float*)d_in[30];
    const float* c1_w  = (const float*)d_in[31];
    const float* c1_b  = (const float*)d_in[32];
    const float* c1_s  = (const float*)d_in[33];
    const float* c1_t  = (const float*)d_in[34];

    char* ws = (char*)d_ws;
    size_t off = 0;
    auto alloc = [&](size_t bytes) { size_t o = off; off = (off + bytes + 255) & ~(size_t)255; return o; };

    float*          seqs = (float*)         (ws + alloc((size_t)NI * Mrow * 64  * 4));  // 16.8 MB
    unsigned short* xz   = (unsigned short*)(ws + alloc((size_t)NI * Mrow * 256 * 2));  // 33.6 MB
    float*          xc   = (float*)         (ws + alloc((size_t)NI * Mrow * 128 * 4));  // 33.6 MB
    float*          dbl  = (float*)         (ws + alloc((size_t)NI * Mrow * 36  * 4));  //  9.4 MB
    float*          yb   = (float*)         (ws + alloc((size_t)NI * Mrow * 128 * 4));  // 33.6 MB
    float*          mm   = (float*)         (ws + alloc((size_t)NI * Mrow * 64  * 4));  // 16.8 MB
    float*          crs  = (float*)         (ws + alloc((size_t)4  * Mrow * 64  * 4));  //  8.4 MB
    float*          h1   = (float*)         (ws + alloc((size_t)4  * Mrow * 64  * 4));  //  8.4 MB
    float*          cat2 = (float*)         (ws + alloc((size_t)4  * Mrow * 64  * 4));  //  8.4 MB
    float*          h3   = (float*)         (ws + alloc((size_t)Mrow * 64 * 4));        //  2.1 MB
    float*          psi  = (float*)         (ws + alloc((size_t)Mrow * 4));
    float*          Prod = (float*)         (ws + alloc((size_t)NCHAIN * NCH * 16 * 4)); // 2 MB
    float*          Hend = (float*)         (ws + alloc((size_t)NCHAIN * NCH * 16 * 4)); // 2 MB
    float*          Hst  = (float*)         (ws + alloc((size_t)NCHAIN * NCH * 16 * 4)); // 2 MB

    const int T = 256;

    // 1) pack g/x -> sequences [8][8192][64]
    {
        long n = (long)NI * Mrow * 64;
        pack_seqs<<<(unsigned)((n + T - 1) / T), T, 0, stream>>>(g, x, seqs);
    }
    // 2) in-projection (WMMA): xz = seqs @ inw^T, N=256, K=64, out bf16
    gemm_wmma_bf16<<<dim3(4, Mrow / 128, NI), T, 0, stream>>>(
        seqs, (long)Mrow * 64, inw, 256L * 64, xz, (long)Mrow * 256,
        256, 64, nullptr, nullptr, nullptr, 0, 0, 1);
    // 3) causal depthwise conv1d + SiLU -> xc
    {
        long n = (long)NI * Mrow * 128;
        dwconv1d_silu<<<(unsigned)((n + T - 1) / T), T, 0, stream>>>(xz, cw, cb, xc);
    }
    // 4) x-projection (WMMA): dbl = xc @ xpw^T, N=36, K=128
    gemm_wmma_bf16<<<dim3(1, Mrow / 128, NI), T, 0, stream>>>(
        xc, (long)Mrow * 128, xpw, 36L * 128, dbl, (long)Mrow * 36,
        36, 128, nullptr, nullptr, nullptr, 0, 0, 0);
    // 5) chunk-parallel selective scan + SiLU(z) gating -> yb
    scan_pass1<<<(NCHAIN * NCH + T - 1) / T, T, 0, stream>>>(
        dbl, xc, dtw, dtb, Alog, Prod, Hend);
    scan_combine<<<(NCHAIN * 16 + T - 1) / T, T, 0, stream>>>(Prod, Hend, Hst);
    scan_pass2<<<(NCHAIN * NCH + T - 1) / T, T, 0, stream>>>(
        dbl, xc, xz, dtw, dtb, Alog, Dpp, Hst, yb);
    // 6) out-projection (WMMA): mm = yb @ ow^T, N=64, K=128
    gemm_wmma_bf16<<<dim3(1, Mrow / 128, NI), T, 0, stream>>>(
        yb, (long)Mrow * 128, ow, 64L * 128, mm, (long)Mrow * 64,
        64, 128, nullptr, nullptr, nullptr, 0, 0, 0);
    // 7) cross gating (two depthwise 3x3 + BN + ReLU + sigmoid mix)
    {
        long n = 4L * Mrow * 64;
        branch_cross<<<(unsigned)((n + T - 1) / T), T, 0, stream>>>(
            mm, dwg_w, dwg_b, dwg_s, dwg_t, dwx_w, dwx_b, dwx_s, dwx_t, crs);
        branch_pdw<<<(unsigned)((n + T - 1) / T), T, 0, stream>>>(
            crs, pdw_w, pdw_b, pdw_s, pdw_t, h1);
    }
    // 8) p1 1x1 conv (WMMA) per branch, fused bias+BN+ReLU -> cat2
    gemm_wmma_bf16<<<dim3(1, Mrow / 128, 4), T, 0, stream>>>(
        h1, (long)Mrow * 64, p1_w, 64L * 64, cat2, (long)Mrow * 64,
        64, 64, p1_b, p1_s, p1_t, 64, 1, 0);
    // 9) c3 3x3 conv 256->64 as implicit WMMA GEMM, fused bias+BN+ReLU
    conv3_wmma_bf16<<<dim3(1, Mrow / 128, 1), T, 0, stream>>>(
        cat2, c3_w, h3, c3_b, c3_s, c3_t);
    // 10) psi = sigmoid(bn(h3 @ c1_w + b))
    psi_kernel<<<(Mrow + T - 1) / T, T, 0, stream>>>(h3, c1_w, c1_b, c1_s, c1_t, psi);
    // 11) out = x * psi
    {
        long n = 2L * 256 * Lh;
        final_mul<<<(unsigned)((n + T - 1) / T), T, 0, stream>>>(x, psi, (float*)d_out);
    }
    (void)in_sizes; (void)n_in; (void)out_size; (void)ws_size;
}
